// SubstructureNeuralNet_44744969290502
// MI455X (gfx1250) — matrix operand order repaired and checked by reading, not compiled, not run on hardware
//
#include <hip/hip_runtime.h>

// Problem constants (match reference)
#define H    128
#define H2   256
#define NN   100000
#define EE   600000
#define LLAY 3
#define SSUB 2
#define OUTD 128
#define NSUB_N 20000
#define ESUB_N 200000
#define BN_EPS 1e-5f

typedef __attribute__((ext_vector_type(16))) _Float16 v16h;
typedef __attribute__((ext_vector_type(8)))  float    v8f;
typedef __attribute__((ext_vector_type(4)))  float    f4;

// All weight panels in this net are K*Nc == 32768 elements (128x256 or 256x128)
#define WPANEL 32768

// ---------------------------------------------------------------------------
// Weight pre-pack: fp32 W[K x Nc] -> f16 fragments laid out exactly as the
// WMMA B operand wants them, so the GEMM hot loop loads each 16-half fragment
// slice with one contiguous 32-byte access.
// Packed index o = (((t*(K/32) + ks)*32 + lane)*16 + e)
//   maps to W[k*Nc + n],  k = ks*32 + (lane>=16 ? 16 : 0) + e,
//                         n = t*16 + (lane & 15)
// ---------------------------------------------------------------------------
__global__ void k_pack(const float* __restrict__ W, _Float16* __restrict__ PW,
                       int K, int Nc) {
  int o = blockIdx.x * blockDim.x + threadIdx.x;
  if (o >= K * Nc) return;
  int e  = o & 15;
  int l  = (o >> 4) & 31;
  int r  = o >> 9;
  int KS = K >> 5;
  int ks = r % KS;
  int t  = r / KS;
  int k  = ks * 32 + ((l & 16) ? 16 : 0) + e;
  int n  = t * 16 + (l & 15);
  PW[o] = (_Float16)W[(size_t)k * Nc + n];
}

// ---------------------------------------------------------------------------
// WMMA GEMM: C[M x Nc] = act(A[M x K] @ W + bias); A fp32 (converted on the
// fly), W pre-packed f16 and staged in LDS once per block, f32 accumulate.
// 4 waves / block; each wave owns NT consecutive 16x16 n-tiles of one m-tile,
// reusing the A fragment across NT WMMAs and reading B fragments from LDS
// (ds_load_b128 pairs). Grid = (M/16, 1); waves exactly tile Nc
// (NT*4*16 == Nc) -> no guards, EXEC all-ones (WMMA requirement).
// ---------------------------------------------------------------------------
template <int NT, bool RELU, bool ACC>
__global__ __launch_bounds__(128)
void gemm_wmma(const float* __restrict__ A, const _Float16* __restrict__ PW,
               const float* __restrict__ bias, float* __restrict__ C,
               int M, int K, int Nc) {
  __shared__ _Float16 lw[WPANEL];   // 64 KB packed weight panel

  // ---- cooperative stage: global -> LDS (stream, 16B granules) ----
  {
    const f4* src = (const f4*)PW;      // 16-byte chunks
    f4*       dst = (f4*)lw;
#pragma unroll 4
    for (int i = threadIdx.x; i < WPANEL / 8; i += 128) dst[i] = src[i];
  }
  __syncthreads();

  const int lane = threadIdx.x & 31;
  const int wave = threadIdx.x >> 5;          // 0..3
  const int m0   = blockIdx.x * 16;
  const int nt0  = wave * NT;                 // first n-tile for this wave

  const int half_id = lane >> 4;              // 0: lanes 0-15, 1: lanes 16-31
  const int lid     = lane & 15;

  // A-fragment addressing: contiguous k order (k = 16*half_id + e), matching packed B
  const int am     = m0 + lid;
  const int a_off0 = half_id ? 16 : 0;
  const int a_off1 = half_id ? 24 : 8;

  const int KS = K >> 5;

  v8f c[NT];
#pragma unroll
  for (int nt = 0; nt < NT; ++nt) {
    const float bv = bias[(nt0 + nt) * 16 + lid];
#pragma unroll
    for (int r = 0; r < 8; ++r) c[nt][r] = bv;
  }

  for (int ks = 0; ks < KS; ++ks) {
    // A fragment (global, fp32 -> f16)
    const float* arow = A + (size_t)am * K + ks * 32;
    f4 a0 = *(const f4*)(arow + a_off0);
    f4 a1 = *(const f4*)(arow + a_off0 + 4);
    f4 a2 = *(const f4*)(arow + a_off1);
    f4 a3 = *(const f4*)(arow + a_off1 + 4);
    v16h af;
#pragma unroll
    for (int i = 0; i < 4; ++i) {
      af[i]      = (_Float16)a0[i];
      af[4 + i]  = (_Float16)a1[i];
      af[8 + i]  = (_Float16)a2[i];
      af[12 + i] = (_Float16)a3[i];
    }
    // All NT B fragments first (ds loads issue back-to-back), then WMMA burst
    v16h bf[NT];
#pragma unroll
    for (int nt = 0; nt < NT; ++nt)
      bf[nt] = *(const v16h*)&lw[(((nt0 + nt) * KS + ks) * 32 + lane) << 4];
#pragma unroll
    for (int nt = 0; nt < NT; ++nt)
      c[nt] = __builtin_amdgcn_wmma_f32_16x16x32_f16(false, af, false, bf[nt],
                                                     (short)0, c[nt], false, false);
  }

#pragma unroll
  for (int nt = 0; nt < NT; ++nt) {
    const int n = (nt0 + nt) * 16 + lid;
#pragma unroll
    for (int r = 0; r < 8; ++r) {
      const int m = m0 + half_id * 8 + r;
      float v = c[nt][r];
      if (RELU) v = v > 0.f ? v : 0.f;
      const size_t idx = (size_t)m * Nc + n;
      C[idx] = ACC ? (C[idx] + v) : v;
    }
  }
}

// ---------------------------------------------------------------------------
// Elementwise / scatter helpers
// ---------------------------------------------------------------------------
__global__ void k_zero(float* __restrict__ p, size_t n) {
  size_t i = (size_t)blockIdx.x * blockDim.x + threadIdx.x;
  if (i < n) p[i] = 0.f;
}

// out = (1 + eps[l]) * in
__global__ void k_scale(float* __restrict__ out, const float* __restrict__ in,
                        const float* __restrict__ eps, int l, size_t n) {
  size_t i = (size_t)blockIdx.x * blockDim.x + threadIdx.x;
  if (i >= n) return;
  out[i] = (1.f + eps[l]) * in[i];
}

// out[sidx[e]] += in[gidx[e]]  (rows of width H), float atomics
__global__ void k_scatter_add(float* __restrict__ out, const float* __restrict__ in,
                              const int* __restrict__ gidx, const int* __restrict__ sidx,
                              size_t nE) {
  size_t i = (size_t)blockIdx.x * blockDim.x + threadIdx.x;
  if (i >= nE * (size_t)H) return;
  size_t e = i >> 7;          // / H
  int    c = (int)(i & 127);  // % H
  atomicAdd(out + (size_t)sidx[e] * H + c, in[(size_t)gidx[e] * H + c]);
}

// Per-column partial sums / sums-of-squares over a 128-row stripe.
__global__ __launch_bounds__(256)
void k_bn_partial(const float* __restrict__ t, float* __restrict__ stats, int M) {
  const int col = threadIdx.x;        // 256 columns
  const int r0  = blockIdx.x * 128;
  const int r1  = min(r0 + 128, M);
  float s = 0.f, s2 = 0.f;
  for (int r = r0; r < r1; ++r) {
    float v = t[(size_t)r * H2 + col];
    s += v; s2 += v * v;
  }
  atomicAdd(&stats[col], s);
  atomicAdd(&stats[H2 + col], s2);
}

// t = relu((t - mean) * rsqrt(var + eps) * g + b)
__global__ void k_bn_relu(float* __restrict__ t, const float* __restrict__ stats,
                          const float* __restrict__ g, const float* __restrict__ b,
                          int l, float invM, size_t n) {
  size_t i = (size_t)blockIdx.x * blockDim.x + threadIdx.x;
  if (i >= n) return;
  int col = (int)(i & 255);  // % H2
  float m  = stats[col] * invM;
  float vv = stats[H2 + col] * invM - m * m;
  float sc = g[l * H2 + col] * rsqrtf(vv + BN_EPS);
  float val = (t[i] - m) * sc + b[l * H2 + col];
  t[i] = val > 0.f ? val : 0.f;
}

// ---------------------------------------------------------------------------
// Host-side orchestration
// ---------------------------------------------------------------------------
static inline dim3 blks(size_t n) { return dim3((unsigned)((n + 255) / 256)); }

static inline void launch_gemm(bool relu, bool acc, const float* A,
                               const _Float16* PW, const float* bias, float* C,
                               int M, int K, int Nc, hipStream_t s) {
  dim3 grid(M / 16, 1);
  if (Nc == 256) {
    if (relu) gemm_wmma<4, true,  false><<<grid, 128, 0, s>>>(A, PW, bias, C, M, K, Nc);
    else      gemm_wmma<4, false, false><<<grid, 128, 0, s>>>(A, PW, bias, C, M, K, Nc);
  } else {
    if (acc)  gemm_wmma<2, false, true ><<<grid, 128, 0, s>>>(A, PW, bias, C, M, K, Nc);
    else      gemm_wmma<2, false, false><<<grid, 128, 0, s>>>(A, PW, bias, C, M, K, Nc);
  }
}

extern "C" void kernel_launch(void* const* d_in, const int* in_sizes, int n_in,
                              void* d_out, int out_size, void* d_ws, size_t ws_size,
                              hipStream_t stream) {
  const float* x      = (const float*)d_in[0];
  const int*   ei     = (const int*)d_in[1];   // [2, E]
  const int*   se0    = (const int*)d_in[2];   // [2, ESUB]
  const int*   se1    = (const int*)d_in[3];
  const float* msg_w1 = (const float*)d_in[4];
  const float* msg_b1 = (const float*)d_in[5];
  const float* bn_g   = (const float*)d_in[6];
  const float* bn_b   = (const float*)d_in[7];
  const float* msg_w2 = (const float*)d_in[8];
  const float* msg_b2 = (const float*)d_in[9];
  const float* eps_g  = (const float*)d_in[10];
  const float* n2s_w1 = (const float*)d_in[11];
  const float* n2s_b1 = (const float*)d_in[12];
  const float* n2s_w2 = (const float*)d_in[13];
  const float* n2s_b2 = (const float*)d_in[14];
  const float* s2n_w1 = (const float*)d_in[15];
  const float* s2n_b1 = (const float*)d_in[16];
  const float* s2n_w2 = (const float*)d_in[17];
  const float* s2n_b2 = (const float*)d_in[18];
  const float* out_w1 = (const float*)d_in[19];
  const float* out_b1 = (const float*)d_in[20];
  const float* out_w2 = (const float*)d_in[21];
  const float* out_b2 = (const float*)d_in[22];

  // Workspace carve-up
  float* xb    = (float*)d_ws;                    // N x H   current features
  float* hb    = xb  + (size_t)NN * H;            // N x H   GIN agg / msg buffer
  float* tb    = hb  + (size_t)NN * H;            // N x H2  hidden activations
  float* sxb   = tb  + (size_t)NN * H2;           // NSUB x H
  float* stb   = sxb + (size_t)NSUB_N * H;        // NSUB x H2
  float* sxo   = stb + (size_t)NSUB_N * H2;       // NSUB x H
  float* stats = sxo + (size_t)NSUB_N * H;        // 2 x H2
  _Float16* pws = (_Float16*)(stats + 2 * H2);    // packed weights (~2 MB)

  // ---- pack all weight matrices into WMMA B-fragment layout (f16) ----
  size_t po = 0;
  auto pack = [&](const float* W, int K, int Nc) -> _Float16* {
    _Float16* dst = pws + po;
    po += (size_t)K * Nc;
    k_pack<<<blks((size_t)K * Nc), 256, 0, stream>>>(W, dst, K, Nc);
    return dst;
  };
  _Float16 *Pmsg1[LLAY], *Pmsg2[LLAY];
  _Float16 *Pn2s1[LLAY * SSUB], *Pn2s2[LLAY * SSUB];
  _Float16 *Ps2n1[LLAY * SSUB], *Ps2n2[LLAY * SSUB];
  for (int l = 0; l < LLAY; ++l) {
    Pmsg1[l] = pack(msg_w1 + (size_t)l * H * H2, H, H2);
    Pmsg2[l] = pack(msg_w2 + (size_t)l * H2 * H, H2, H);
  }
  for (int i = 0; i < LLAY * SSUB; ++i) {
    Pn2s1[i] = pack(n2s_w1 + (size_t)i * H * H2, H, H2);
    Pn2s2[i] = pack(n2s_w2 + (size_t)i * H2 * H, H2, H);
    Ps2n1[i] = pack(s2n_w1 + (size_t)i * H * H2, H, H2);
    Ps2n2[i] = pack(s2n_w2 + (size_t)i * H2 * H, H2, H);
  }
  _Float16* Pout1 = pack(out_w1, H, H2);
  _Float16* Pout2 = pack(out_w2, H2, OUTD);

  hipMemcpyAsync(xb, x, (size_t)NN * H * sizeof(float),
                 hipMemcpyDeviceToDevice, stream);

  const size_t nNH  = (size_t)NN * H;
  const size_t nNH2 = (size_t)NN * H2;

  for (int l = 0; l < LLAY; ++l) {
    // ---- GINConv: h = (1+eps)*x + segment_sum(x[src] -> dst) ----
    k_scale<<<blks(nNH), 256, 0, stream>>>(hb, xb, eps_g, l, nNH);
    k_scatter_add<<<blks((size_t)EE * H), 256, 0, stream>>>(hb, xb, ei, ei + EE, EE);
    // Linear(H,2H)
    launch_gemm(false, false, hb, Pmsg1[l], msg_b1 + (size_t)l * H2, tb, NN, H, H2, stream);
    // BatchNorm (training batch stats) + ReLU
    k_zero<<<blks(2 * H2), 256, 0, stream>>>(stats, 2 * H2);
    k_bn_partial<<<dim3((NN + 127) / 128), 256, 0, stream>>>(tb, stats, NN);
    k_bn_relu<<<blks(nNH2), 256, 0, stream>>>(tb, stats, bn_g, bn_b, l, 1.f / NN, nNH2);
    // Linear(2H,H) -> new x
    launch_gemm(false, false, tb, Pmsg2[l], msg_b2 + (size_t)l * H, xb, NN, H2, H, stream);

    // ---- substructure message passing ----
    for (int s = 0; s < SSUB; ++s) {
      const int* row = (s == 0) ? se0 : se1;          // node index
      const int* col = row + ESUB_N;                  // substructure index
      const size_t wi = (size_t)(l * SSUB + s);

      // sx = segment_sum(x[row] -> col), then n2s MLP
      k_zero<<<blks((size_t)NSUB_N * H), 256, 0, stream>>>(sxb, (size_t)NSUB_N * H);
      k_scatter_add<<<blks((size_t)ESUB_N * H), 256, 0, stream>>>(sxb, xb, row, col, ESUB_N);
      launch_gemm(true, false, sxb, Pn2s1[wi], n2s_b1 + wi * H2, stb, NSUB_N, H, H2, stream);
      launch_gemm(false, false, stb, Pn2s2[wi], n2s_b2 + wi * H, sxo, NSUB_N, H2, H, stream);

      // msg = segment_sum(sx[col] -> row), then x += s2n MLP(msg)
      k_zero<<<blks(nNH), 256, 0, stream>>>(hb, nNH);
      k_scatter_add<<<blks((size_t)ESUB_N * H), 256, 0, stream>>>(hb, sxo, col, row, ESUB_N);
      launch_gemm(true, false, hb, Ps2n1[wi], s2n_b1 + wi * H2, tb, NN, H, H2, stream);
      launch_gemm(false, true, tb, Ps2n2[wi], s2n_b2 + wi * H, xb, NN, H2, H, stream);
    }
  }

  // ---- output head: relu(x@w1+b1)@w2+b2 ----
  launch_gemm(true, false, xb, Pout1, out_b1, tb, NN, H, H2, stream);
  launch_gemm(false, false, tb, Pout2, out_b2, (float*)d_out, NN, H2, OUTD, stream);
}
